// IndexGumbelVectorQuantizer_2156073582764
// MI455X (gfx1250) — matrix operand order repaired and checked by reading, not compiled, not run on hardware
//
#include <hip/hip_runtime.h>
#include <hip/hip_bf16.h>

typedef __attribute__((ext_vector_type(16))) __bf16 v16bf;
typedef __attribute__((ext_vector_type(8)))  __bf16 v8bf;
typedef __attribute__((ext_vector_type(8)))  float  v8f;

#define B_TOT 65536
#define Hd    512
#define Gg    2
#define Vv    320
#define GVn   640
#define Ee    256
#define MT    32          // batch rows per block
#define LSTRIDE 644       // padded logits row stride (floats) to break LDS bank conflicts

// workspace layout (bytes)
#define WS_COUNTS 0                       // 640 f32 marginal accumulators
#define WS_WHI    4096                    // 640*512 bf16 (hi plane)
#define WS_WLO    (4096 + GVn*Hd*2)       // 640*512 bf16 (lo plane)

__device__ __forceinline__ unsigned short f2bf(float f) {
  unsigned u = __float_as_uint(f);
  u += 0x7fffu + ((u >> 16) & 1u);        // round-to-nearest-even
  return (unsigned short)(u >> 16);
}
__device__ __forceinline__ float bf2f(unsigned short h) {
  return __uint_as_float(((unsigned)h) << 16);
}

// ---------------- kernel 0: W -> bf16 hi/lo planes, zero marginal counters ---------------
__global__ void prep_kernel(const float* __restrict__ W,
                            unsigned short* __restrict__ whi,
                            unsigned short* __restrict__ wlo,
                            float* __restrict__ counts) {
  int i = blockIdx.x * blockDim.x + threadIdx.x;
  if (i < GVn) counts[i] = 0.f;
  if (i < GVn * Hd) {
    float w = W[i];
    unsigned short h = f2bf(w);
    whi[i] = h;
    wlo[i] = f2bf(w - bf2f(h));
  }
}

// ---------------- kernel 1: fused GEMM + gumbel-softmax + gather ----------------
__global__ void __launch_bounds__(256)
main_kernel(const float* __restrict__ x, const float* __restrict__ noise,
            const float* __restrict__ emb, const float* __restrict__ bproj,
            const unsigned short* __restrict__ whi_u, const unsigned short* __restrict__ wlo_u,
            float* __restrict__ counts, float* __restrict__ out) {
  extern __shared__ char smem[];
  unsigned short* Ahi = (unsigned short*)smem;          // 32x512 bf16 hi
  unsigned short* Alo = Ahi + MT * Hd;                  // 32x512 bf16 lo
  float* Llog = (float*)(smem + 2 * MT * Hd * 2);       // 32 x 644 f32 logits (offset 65536)
  char*  redb = smem + 65536 + MT * LSTRIDE * 4;
  float* redM = (float*)redb;                           // 256 partial maxima
  float* redS = redM + 256;                             // 256 partial sums
  int*   redI = (int*)(redS + 256);                     // 256 partial argmax
  int*   pidx = (int*)(redI + 256);                     // 64 merged argmax
  float* pgd  = (float*)(pidx + 64);                    // 64 merged (1-s)+s

  const int tid = threadIdx.x;
  const int rowBase = blockIdx.x * MT;

  // ---- stage x tile as bf16 hi/lo planes in LDS ----
  const float4* xg = (const float4*)(x + (size_t)rowBase * Hd);
  #pragma unroll
  for (int it = 0; it < (MT * Hd / 4) / 256; ++it) {
    int t = tid + it * 256;
    float4 v = xg[t];
    float vv[4] = {v.x, v.y, v.z, v.w};
    int base = t * 4;
    #pragma unroll
    for (int k = 0; k < 4; ++k) {
      unsigned short h = f2bf(vv[k]);
      Ahi[base + k] = h;
      Alo[base + k] = f2bf(vv[k] - bf2f(h));
    }
  }
  __syncthreads();

  const int w    = tid >> 5;      // wave id 0..7 (wave32)
  const int lane = tid & 31;
  const int rr   = lane & 15;
  const int hh   = lane >> 4;

  v8f acc[2][5];
  #pragma unroll
  for (int m = 0; m < 2; ++m)
    #pragma unroll
    for (int j = 0; j < 5; ++j) acc[m][j] = (v8f)0.f;

  const __bf16* AhiB = (const __bf16*)Ahi;
  const __bf16* AloB = (const __bf16*)Alo;
  const __bf16* WhiB = (const __bf16*)whi_u;
  const __bf16* WloB = (const __bf16*)wlo_u;

  for (int c = 0; c < Hd / 32; ++c) {
    // A fragments: lane(h,rr) holds row M=rr, K = c*32 + {h*16..h*16+15}
    v16bf ah[2], al[2];
    #pragma unroll
    for (int m = 0; m < 2; ++m) {
      int row = m * 16 + rr;
      const __bf16* p = AhiB + row * Hd + c * 32 + hh * 16;
      v8bf x1 = *(const v8bf*)p;
      v8bf x2 = *(const v8bf*)(p + 8);
      ah[m] = __builtin_shufflevector(x1, x2, 0,1,2,3,4,5,6,7,8,9,10,11,12,13,14,15);
      const __bf16* q = AloB + row * Hd + c * 32 + hh * 16;
      v8bf y1 = *(const v8bf*)q;
      v8bf y2 = *(const v8bf*)(q + 8);
      al[m] = __builtin_shufflevector(y1, y2, 0,1,2,3,4,5,6,7,8,9,10,11,12,13,14,15);
    }
    #pragma unroll
    for (int j = 0; j < 5; ++j) {
      int n = (w * 5 + j) * 16 + rr;              // W row (output column)
      size_t boff = (size_t)n * Hd + c * 32 + hh * 16;  // 16 contiguous K per lane
      __builtin_prefetch(WhiB + boff + 32, 0, 1); // next K-chunk into L2
      v16bf bh = *(const v16bf*)(WhiB + boff);
      v16bf bl = *(const v16bf*)(WloB + boff);
      #pragma unroll
      for (int m = 0; m < 2; ++m) {
        acc[m][j] = __builtin_amdgcn_wmma_f32_16x16x32_bf16(false, ah[m], false, bh, (short)0, acc[m][j], false, false);
        acc[m][j] = __builtin_amdgcn_wmma_f32_16x16x32_bf16(false, al[m], false, bh, (short)0, acc[m][j], false, false);
        acc[m][j] = __builtin_amdgcn_wmma_f32_16x16x32_bf16(false, ah[m], false, bl, (short)0, acc[m][j], false, false);
      }
    }
  }

  // ---- logits + bias into padded LDS (C layout: elem i -> M = i + 8*h, N = rr) ----
  #pragma unroll
  for (int j = 0; j < 5; ++j) {
    int col = (w * 5 + j) * 16 + rr;
    float bj = bproj[col];
    #pragma unroll
    for (int m = 0; m < 2; ++m)
      #pragma unroll
      for (int i = 0; i < 8; ++i)
        Llog[(m * 16 + hh * 8 + i) * LSTRIDE + col] = acc[m][j][i] + bj;
  }
  __syncthreads();

  // ---- online gumbel-softmax: 4 threads per (row,g), 80 v's each ----
  {
    int p = tid >> 2;      // pair 0..63
    int q = tid & 3;
    int r = p >> 1;
    int g = p & 1;
    size_t b = (size_t)(rowBase + r);
    const float* urow = noise + (b * Gg + g) * Vv;
    const float* lrow = Llog + r * LSTRIDE + g * Vv;
    float mx = -3.4e38f, sum = 0.f; int idx = -1;
    for (int v = q * 80; v < q * 80 + 80; ++v) {
      float u = urow[v];
      float gum = -__logf(-__logf(u + 1e-10f) + 1e-10f);
      float z = (lrow[v] + gum) * 0.5f;                 // / TAU
      if (z > mx) { sum = sum * __expf(mx - z) + 1.f; mx = z; idx = v; }
      else        { sum += __expf(z - mx); }
    }
    redM[tid] = mx; redS[tid] = sum; redI[tid] = idx;
  }
  __syncthreads();
  if ((tid & 3) == 0) {
    int p = tid >> 2;
    float M = redM[tid], S = redS[tid]; int I = redI[tid];
    #pragma unroll
    for (int q = 1; q < 4; ++q) {     // earlier quarter wins ties -> first-index argmax
      float mq = redM[tid + q], sq = redS[tid + q]; int iq = redI[tid + q];
      if (mq > M) { S = S * __expf(M - mq) + sq; M = mq; I = iq; }
      else        { S += sq * __expf(mq - M); }
    }
    float smax = 1.f / S;             // y_soft at argmax
    float gd = (1.f - smax) + smax;   // matches y_hard - sg(y_soft) + y_soft in fp32
    atomicAdd(&counts[(p & 1) * Vv + I], gd);
    pidx[p] = I; pgd[p] = gd;
  }
  __syncthreads();

  // ---- selected = gd * embeddings[g, idx, :], coalesced 1KB stores ----
  for (int p = 0; p < 64; ++p) {
    int r = p >> 1, g = p & 1;
    size_t b = (size_t)(rowBase + r);
    float val = pgd[p] * emb[(size_t)(g * Vv + pidx[p]) * Ee + tid];
    out[(b * Gg + g) * Ee + tid] = val;
  }
}

// ---------------- kernel 2: perplexity from marginal counts ----------------
__global__ void ppl_kernel(const float* __restrict__ counts, float* __restrict__ out) {
  __shared__ float s0[Vv], s1[Vv];
  int v = threadIdx.x;
  const float inv = 1.f / 65536.f;
  float m0 = counts[v] * inv;
  float m1 = counts[Vv + v] * inv;
  s0[v] = m0 * __logf(m0 + 1e-7f);
  s1[v] = m1 * __logf(m1 + 1e-7f);
  __syncthreads();
  if (v == 0) {
    float a = 0.f, bq = 0.f;
    for (int i = 0; i < Vv; ++i) { a += s0[i]; bq += s1[i]; }
    out[(size_t)B_TOT * Gg * Ee] = __expf(-a) + __expf(-bq);
  }
}

extern "C" void kernel_launch(void* const* d_in, const int* in_sizes, int n_in,
                              void* d_out, int out_size, void* d_ws, size_t ws_size,
                              hipStream_t stream) {
  const float* x     = (const float*)d_in[0];   // (B,1,H)
  const float* noise = (const float*)d_in[1];   // (B,G,V)
  const float* emb   = (const float*)d_in[2];   // (G,V,E)
  const float* W     = (const float*)d_in[3];   // (G*V,H)
  const float* bproj = (const float*)d_in[4];   // (G*V,)
  float* out = (float*)d_out;

  float* counts = (float*)((char*)d_ws + WS_COUNTS);
  unsigned short* whi = (unsigned short*)((char*)d_ws + WS_WHI);
  unsigned short* wlo = (unsigned short*)((char*)d_ws + WS_WLO);

  prep_kernel<<<(GVn * Hd + 255) / 256, 256, 0, stream>>>(W, whi, wlo, counts);

  size_t smem = 2 * MT * Hd * 2          // A hi/lo planes (64 KB)
              + MT * LSTRIDE * 4         // logits (82.4 KB)
              + 256 * 4 * 3              // reduction scratch
              + 64 * 4 * 2;              // merged idx/gd
  main_kernel<<<B_TOT / MT, 256, smem, stream>>>(x, noise, emb, bproj, whi, wlo, counts, out);

  ppl_kernel<<<1, Vv, 0, stream>>>(counts, out);
}
